// CorrNet_24429773979764
// MI455X (gfx1250) — matrix operand order repaired and checked by reading, not compiled, not run on hardware
//
#include <hip/hip_runtime.h>
#include <hip/hip_bf16.h>
#include <math.h>

typedef __attribute__((ext_vector_type(16))) __bf16 v16bf;
typedef __attribute__((ext_vector_type(8)))  float  v8f;

#define WAVES_PER_BLOCK 4
#define NEG_SLOPE 0.2f
#define GN_EPS 1e-5f
#define NPTS 8192

__device__ __forceinline__ float lrelu(float x) { return x > 0.f ? x : NEG_SLOPE * x; }

// Pack two contiguous 8-float runs into a bf16 A/B fragment register (16 elems).
__device__ __forceinline__ v16bf pack16(const float* __restrict__ p1,
                                        const float* __restrict__ p2) {
    const float4 q0 = *(const float4*)(p1);
    const float4 q1 = *(const float4*)(p1 + 4);
    const float4 q2 = *(const float4*)(p2);
    const float4 q3 = *(const float4*)(p2 + 4);
    v16bf a;
    a[0]  = (__bf16)q0.x; a[1]  = (__bf16)q0.y; a[2]  = (__bf16)q0.z; a[3]  = (__bf16)q0.w;
    a[4]  = (__bf16)q1.x; a[5]  = (__bf16)q1.y; a[6]  = (__bf16)q1.z; a[7]  = (__bf16)q1.w;
    a[8]  = (__bf16)q2.x; a[9]  = (__bf16)q2.y; a[10] = (__bf16)q2.z; a[11] = (__bf16)q2.w;
    a[12] = (__bf16)q3.x; a[13] = (__bf16)q3.y; a[14] = (__bf16)q3.z; a[15] = (__bf16)q3.w;
    return a;
}

// Y = [FUSE? GroupNorm(lrelu(X @ Wp^T + b)) : X @ Wp^T + b]
// X: [N, CINP] f32, zero-padded rows (CINP multiple of 32).
// Wp: [COUT, CINP] bf16, zero-padded (pre-packed).
// One wave computes 16 rows x COUT. A-fragments preloaded once, reused per N-tile.
// Two accumulators per N-tile shorten the WMMA D->C dependency chain.
template<int CINP, int COUT, bool FUSE>
__global__ __launch_bounds__(128) void linear_wmma_kernel(
    const float* __restrict__ X, const __bf16* __restrict__ Wp,
    const float* __restrict__ Bv, const float* __restrict__ GW,
    const float* __restrict__ GB, float* __restrict__ Y)
{
    __shared__ __align__(16) float smem[WAVES_PER_BLOCK * 16 * 132];
    const int lane   = threadIdx.x & 31;
    const int wave   = threadIdx.x >> 5;
    const int laneHi = lane >> 4;
    const int lane15 = lane & 15;
    const int rowBase = (blockIdx.x * WAVES_PER_BLOCK + wave) * 16;
    float* ssm = smem + wave * 16 * 132;

    constexpr int NKT = CINP / 32;
    constexpr int NNT = COUT / 16;

    // Preload all A-fragments for this wave's 16 rows (branch-free float4 loads).
    // A layout: lane15 = row, element e -> k = kbase + 16*laneHi + e
    const float* xrow = X + (size_t)(rowBase + lane15) * CINP;
    v16bf afr[NKT];
#pragma unroll
    for (int kt = 0; kt < NKT; ++kt) {
        const float* p = xrow + kt * 32 + 16 * laneHi;
        afr[kt] = pack16(p, p + 8);
    }

#pragma unroll
    for (int nt = 0; nt < NNT; ++nt) {
        v8f acc0 = {}, acc1 = {};
        // B layout: lane15 = out-channel, element e -> k = kbase + 16*laneHi + e
        const __bf16* wrow = Wp + (size_t)(nt * 16 + lane15) * CINP + 16 * laneHi;
#pragma unroll
        for (int kt = 0; kt < NKT; kt += 2) {
            const v16bf bfr0 = *(const v16bf*)(wrow + kt * 32);
            acc0 = __builtin_amdgcn_wmma_f32_16x16x32_bf16(
                false, afr[kt], false, bfr0, (short)0, acc0, false, false);
            if (kt + 1 < NKT) {
                const v16bf bfr1 = *(const v16bf*)(wrow + (kt + 1) * 32);
                acc1 = __builtin_amdgcn_wmma_f32_16x16x32_bf16(
                    false, afr[kt + 1], false, bfr1, (short)0, acc1, false, false);
            }
        }
        // D layout: acc[v] -> row (v + 8*laneHi), col lane15
        const int n = nt * 16 + lane15;
        const float bias = Bv[n];
#pragma unroll
        for (int v = 0; v < 8; ++v) {
            const int r = v + 8 * laneHi;
            const float val = acc0[v] + acc1[v] + bias;
            if (FUSE) {
                ssm[r * 132 + n] = lrelu(val);
            } else {
                Y[(size_t)(rowBase + r) * COUT + n] = val;
            }
        }
    }
    if (!FUSE) return;
    __syncthreads();
    // GroupNorm over contiguous 32-channel groups; lanes 0..15 own one row each.
    if (lane < 16) {
        const int row = rowBase + lane;
        const float* src = ssm + lane * 132;
#pragma unroll
        for (int g = 0; g < COUT / 32; ++g) {
            const int base = g * 32;
            float s = 0.f;
#pragma unroll
            for (int j = 0; j < 32; ++j) s += src[base + j];
            const float mu = s * (1.f / 32.f);
            float v2 = 0.f;
#pragma unroll
            for (int j = 0; j < 32; ++j) { float d = src[base + j] - mu; v2 += d * d; }
            const float inv = rsqrtf(v2 * (1.f / 32.f) + GN_EPS);
#pragma unroll
            for (int j = 0; j < 32; ++j) {
                const int c = base + j;
                Y[(size_t)row * COUT + c] = (src[c] - mu) * inv * GW[c] + GB[c];
            }
        }
    }
}

// vtx/pts [N,3] -> [N,32] f32 zero-padded
__global__ void pack_input_kernel(const float* __restrict__ x, float* __restrict__ xp, int N)
{
    const int i = blockIdx.x * blockDim.x + threadIdx.x;
    if (i >= N * 32) return;
    const int r = i >> 5, c = i & 31;
    xp[i] = (c < 3) ? x[r * 3 + c] : 0.f;
}

// W [Cout,Cin] f32 -> Wp [Cout,Kpad] bf16 zero-padded
__global__ void pack_weight_kernel(const float* __restrict__ W, __bf16* __restrict__ Wp,
                                   int Cout, int Cin, int Kpad)
{
    const int i = blockIdx.x * blockDim.x + threadIdx.x;
    if (i >= Cout * Kpad) return;
    const int n = i / Kpad, k = i % Kpad;
    Wp[i] = (__bf16)((k < Cin) ? W[n * Cin + k] : 0.f);
}

// out[c] = max over N points of X[:, c]; one block per channel
__global__ __launch_bounds__(256) void colmax_kernel(
    const float* __restrict__ X, float* __restrict__ out, int N, int C)
{
    const int c = blockIdx.x;
    float m = -3.4e38f;
    for (int i = threadIdx.x; i < N; i += 256) m = fmaxf(m, X[(size_t)i * C + c]);
    __shared__ float red[8];
    for (int off = 16; off > 0; off >>= 1) m = fmaxf(m, __shfl_xor(m, off));
    if ((threadIdx.x & 31) == 0) red[threadIdx.x >> 5] = m;
    __syncthreads();
    if (threadIdx.x == 0) {
        float r = red[0];
        for (int w = 1; w < 8; ++w) r = fmaxf(r, red[w]);
        out[c] = r;
    }
}

// x4[i, 0:128] = x3 (broadcast), x4[i, 128:256] = x1[i]
__global__ void concat_kernel(const float* __restrict__ x3, const float* __restrict__ x1,
                              float* __restrict__ x4, int N)
{
    const int i = blockIdx.x * blockDim.x + threadIdx.x;
    if (i >= N * 256) return;
    const int r = i >> 8, c = i & 255;
    x4[i] = (c < 128) ? x3[c] : x1[(size_t)r * 128 + (c - 128)];
}

// row-wise L2 normalize 32-channel rows; one wave per row
__global__ __launch_bounds__(256) void l2norm_kernel(
    const float* __restrict__ X, float* __restrict__ Y, int N)
{
    const int row  = blockIdx.x * 8 + (threadIdx.x >> 5);
    const int lane = threadIdx.x & 31;
    if (row >= N) return;
    const float v = X[(size_t)row * 32 + lane];
    float ss = v * v;
    for (int off = 16; off > 0; off >>= 1) ss += __shfl_xor(ss, off);
    const float nrm = fmaxf(sqrtf(ss), 1e-12f);
    Y[(size_t)row * 32 + lane] = v / nrm;
}

__device__ __forceinline__ v16bf load_bfrag32(const float* __restrict__ brow) {
    const float4 q0 = *(const float4*)(brow);
    const float4 q1 = *(const float4*)(brow + 4);
    const float4 q2 = *(const float4*)(brow + 8);
    const float4 q3 = *(const float4*)(brow + 12);
    v16bf bm;
    bm[0]  = (__bf16)q0.x; bm[1]  = (__bf16)q0.y; bm[2]  = (__bf16)q0.z; bm[3]  = (__bf16)q0.w;
    bm[4]  = (__bf16)q1.x; bm[5]  = (__bf16)q1.y; bm[6]  = (__bf16)q1.z; bm[7]  = (__bf16)q1.w;
    bm[8]  = (__bf16)q2.x; bm[9]  = (__bf16)q2.y; bm[10] = (__bf16)q2.z; bm[11] = (__bf16)q2.w;
    bm[12] = (__bf16)q3.x; bm[13] = (__bf16)q3.y; bm[14] = (__bf16)q3.z; bm[15] = (__bf16)q3.w;
    return bm;
}

// Fused cos-sim (A[N,32] @ B[M,32]^T) + row max/argmax, no NxM materialization.
// jt loop processed 2 tiles per iteration with independent accumulators so the
// bf16-WMMA->VALU hazard slots of one tile are filled by the other tile's work.
__global__ __launch_bounds__(128) void cossim_argmax_kernel(
    const float* __restrict__ A, const float* __restrict__ B,
    float* __restrict__ sim, int* __restrict__ idx, int N, int M)
{
    const int lane   = threadIdx.x & 31;
    const int wave   = threadIdx.x >> 5;
    const int laneHi = lane >> 4;
    const int lane15 = lane & 15;
    const int rowBase = (blockIdx.x * 4 + wave) * 16;

    const float* arow = A + (size_t)(rowBase + lane15) * 32 + 16 * laneHi;
    const v16bf a = pack16(arow, arow + 8);

    float best[8]; int bidx[8];
#pragma unroll
    for (int v = 0; v < 8; ++v) { best[v] = -3.4e38f; bidx[v] = 0; }

    const int nJT = M >> 4;
    for (int jt = 0; jt < nJT; jt += 2) {
        const float* brow0 = B + (size_t)(jt * 16 + lane15) * 32 + 16 * laneHi;
        const float* brow1 = brow0 + 16 * 32;
        const v16bf bm0 = load_bfrag32(brow0);
        const v16bf bm1 = load_bfrag32(brow1);
        v8f acc0 = {}, acc1 = {};
        acc0 = __builtin_amdgcn_wmma_f32_16x16x32_bf16(
            false, a, false, bm0, (short)0, acc0, false, false);
        acc1 = __builtin_amdgcn_wmma_f32_16x16x32_bf16(
            false, a, false, bm1, (short)0, acc1, false, false);
        const int j0 = jt * 16 + lane15;
        const int j1 = j0 + 16;
#pragma unroll
        for (int v = 0; v < 8; ++v)
            if (acc0[v] > best[v]) { best[v] = acc0[v]; bidx[v] = j0; }
#pragma unroll
        for (int v = 0; v < 8; ++v)
            if (acc1[v] > best[v]) { best[v] = acc1[v]; bidx[v] = j1; }
    }
#pragma unroll
    for (int v = 0; v < 8; ++v) {
        float bm = best[v]; int bi = bidx[v];
        for (int off = 1; off < 16; off <<= 1) {
            const float om = __shfl_xor(bm, off);
            const int   oi = __shfl_xor(bi, off);
            if (om > bm || (om == bm && oi < bi)) { bm = om; bi = oi; }
        }
        if (lane15 == 0) {
            const int r = rowBase + v + 8 * laneHi;
            sim[r] = bm;
            idx[r] = bi;
        }
    }
}

// ocin[i] = [out_vtx[i,0:32], out_pts[idx[i],0:32], sim[i], 0-pad] (stride 96)
__global__ void build_ocin_kernel(const float* __restrict__ ov, const float* __restrict__ op,
                                  const float* __restrict__ sim, const int* __restrict__ idx,
                                  float* __restrict__ ocin, int N)
{
    const int i = blockIdx.x * blockDim.x + threadIdx.x;
    if (i >= N) return;
    const float* a = ov + (size_t)i * 32;
    const float* b = op + (size_t)idx[i] * 32;
    float* o = ocin + (size_t)i * 96;
    for (int c = 0; c < 32; ++c) o[c] = a[c];
    for (int c = 0; c < 32; ++c) o[32 + c] = b[c];
    o[64] = sim[i];
    for (int c = 65; c < 96; ++c) o[c] = 0.f;
}

// oc[i] = dot(X[i,0:64], w) + b ; one wave per row
__global__ __launch_bounds__(256) void clin_kernel(
    const float* __restrict__ X, const float* __restrict__ w, const float* __restrict__ b,
    float* __restrict__ Y, int N)
{
    const int row  = blockIdx.x * 8 + (threadIdx.x >> 5);
    const int lane = threadIdx.x & 31;
    if (row >= N) return;
    float s = X[(size_t)row * 64 + lane] * w[lane]
            + X[(size_t)row * 64 + 32 + lane] * w[32 + lane];
    for (int off = 16; off > 0; off >>= 1) s += __shfl_xor(s, off);
    if (lane == 0) Y[row] = s + b[0];
}

extern "C" void kernel_launch(void* const* d_in, const int* in_sizes, int n_in,
                              void* d_out, int out_size, void* d_ws, size_t ws_size,
                              hipStream_t stream) {
    const int N = NPTS;
    const float* vtx = (const float*)d_in[0];
    const float* pts = (const float*)d_in[1];

    // params flattened in insertion order:
    // pn.m1[3x(W,b,gw,gb)], pn.m2[1x], pn.m3[2x], pn.lin_w, pn.lin_b,
    // cm[1x(W,b,gw,gb)], clin_w, clin_b
    const float *Wt[7], *Bt[7], *Gw[7], *Gb[7];
    int p = 2;
    for (int i = 0; i < 6; ++i) {
        Wt[i] = (const float*)d_in[p++]; Bt[i] = (const float*)d_in[p++];
        Gw[i] = (const float*)d_in[p++]; Gb[i] = (const float*)d_in[p++];
    }
    const float* lin_w = (const float*)d_in[p++];
    const float* lin_b = (const float*)d_in[p++];
    Wt[6] = (const float*)d_in[p++]; Bt[6] = (const float*)d_in[p++];
    Gw[6] = (const float*)d_in[p++]; Gb[6] = (const float*)d_in[p++];
    const float* clin_w = (const float*)d_in[p++];
    const float* clin_b = (const float*)d_in[p++];

    // ---- workspace carve-up (~25 MB) ----
    float* xp   = (float*)d_ws;            // [N,32]  padded input
    float* t0   = xp + (size_t)N * 32;     // [N,256]
    float* t1   = t0 + (size_t)N * 256;    // [N,256]
    float* x1b  = t1 + (size_t)N * 256;    // [N,128]
    float* x3b  = x1b + (size_t)N * 128;   // [128]
    float* simb = x3b + 128;               // [N]
    float* ocin = simb + N;                // [N,96]
    int*   idxb = (int*)(ocin + (size_t)N * 96);  // [N]
    // packed bf16 weights (Kpad = Cin rounded up to 32)
    __bf16* wp0   = (__bf16*)(idxb + N);   // 32x32
    __bf16* wp1   = wp0 + 32 * 32;         // 64x32
    __bf16* wp2   = wp1 + 64 * 32;         // 128x64
    __bf16* wp3   = wp2 + 128 * 64;        // 128x128
    __bf16* wp4   = wp3 + 128 * 128;       // 128x256
    __bf16* wp5   = wp4 + 128 * 256;       // 64x128
    __bf16* wplin = wp5 + 64 * 128;        // 32x64
    __bf16* wpcm  = wplin + 32 * 64;       // 64x96

    float* out_vtx = (float*)d_out;
    float* out_pts = out_vtx + (size_t)N * 32;
    float* oc      = out_pts + (size_t)N * 32;

    // ---- pre-pack weights (tiny, every launch; deterministic) ----
    auto packw = [&](const float* W, __bf16* Wp, int Cout, int Cin, int Kpad) {
        const int tot = Cout * Kpad;
        pack_weight_kernel<<<(tot + 255) / 256, 256, 0, stream>>>(W, Wp, Cout, Cin, Kpad);
    };
    packw(Wt[0], wp0, 32, 3, 32);
    packw(Wt[1], wp1, 64, 32, 32);
    packw(Wt[2], wp2, 128, 64, 64);
    packw(Wt[3], wp3, 128, 128, 128);
    packw(Wt[4], wp4, 128, 256, 256);
    packw(Wt[5], wp5, 64, 128, 128);
    packw(lin_w, wplin, 32, 64, 64);
    packw(Wt[6], wpcm, 64, 65, 96);

    const dim3 blk(128);
    const dim3 grd(N / 64);

    auto run_pointnet = [&](const float* x, float* outp) {
        pack_input_kernel<<<(N * 32 + 255) / 256, 256, 0, stream>>>(x, xp, N);
        linear_wmma_kernel< 32,  32, true ><<<grd, blk, 0, stream>>>(xp,  wp0, Bt[0], Gw[0], Gb[0], t0);
        linear_wmma_kernel< 32,  64, true ><<<grd, blk, 0, stream>>>(t0,  wp1, Bt[1], Gw[1], Gb[1], t1);
        linear_wmma_kernel< 64, 128, true ><<<grd, blk, 0, stream>>>(t1,  wp2, Bt[2], Gw[2], Gb[2], x1b);
        linear_wmma_kernel<128, 128, true ><<<grd, blk, 0, stream>>>(x1b, wp3, Bt[3], Gw[3], Gb[3], t0);
        colmax_kernel<<<128, 256, 0, stream>>>(t0, x3b, N, 128);
        concat_kernel<<<(N * 256 + 255) / 256, 256, 0, stream>>>(x3b, x1b, t1, N);
        linear_wmma_kernel<256, 128, true ><<<grd, blk, 0, stream>>>(t1, wp4, Bt[4], Gw[4], Gb[4], t0);
        linear_wmma_kernel<128,  64, true ><<<grd, blk, 0, stream>>>(t0, wp5, Bt[5], Gw[5], Gb[5], t1);
        linear_wmma_kernel< 64,  32, false><<<grd, blk, 0, stream>>>(t1, wplin, lin_b, nullptr, nullptr, t0);
        l2norm_kernel<<<N / 8, 256, 0, stream>>>(t0, outp, N);
    };

    run_pointnet(vtx, out_vtx);
    run_pointnet(pts, out_pts);

    cossim_argmax_kernel<<<N / 64, 128, 0, stream>>>(out_vtx, out_pts, simb, idxb, N, N);
    build_ocin_kernel<<<(N + 255) / 256, 256, 0, stream>>>(out_vtx, out_pts, simb, idxb, ocin, N);
    linear_wmma_kernel<96, 64, true><<<grd, blk, 0, stream>>>(ocin, wpcm, Bt[6], Gw[6], Gb[6], t0);
    clin_kernel<<<N / 8, 256, 0, stream>>>(t0, clin_w, clin_b, oc, N);
}
